// MoELayer_6923487282556
// MI455X (gfx1250) — matrix operand order repaired and checked by reading, not compiled, not run on hardware
//
#include <hip/hip_runtime.h>
#include <math.h>

// MoE top-2 layer for gfx1250 (MI455X), bf16 WMMA path with weight pre-conversion.
// d_in: x[16384,2048] f32, router_w[8,2048], gate_w[8,3072,2048],
//       up_w[8,3072,2048], down_w[8,2048,3072]  (all f32)
// d_out: [16384,2048] f32

#define D_MODEL 2048
#define FFN_H   3072
#define N_EXP   8
#define BM      16   // tokens per workgroup tile (one WMMA M-tile)

typedef __attribute__((ext_vector_type(16))) __bf16 v16bf;
typedef __attribute__((ext_vector_type(2)))  __bf16 v2bf;
typedef __attribute__((ext_vector_type(8)))  float  v8f;

union Frag16 {
  v16bf v;
  unsigned short s[16];
  unsigned u[8];
  uint4 q[2];
};

#if __has_builtin(__builtin_amdgcn_cvt_pk_bf16_f32)
#define HAVE_CVT_PK_BF16 1
#else
#define HAVE_CVT_PK_BF16 0
#endif

// pack two f32 -> two bf16 in one dword.
__device__ __forceinline__ unsigned pack2bf(float a, float b) {
#if HAVE_CVT_PK_BF16
  union { v2bf v; unsigned u; } r;
  r.v = __builtin_amdgcn_cvt_pk_bf16_f32(a, b);
  return r.u;
#else
  union { float f; unsigned u; } x, y;
  x.f = a; y.f = b;
  unsigned au = x.u + 0x8000u;   // round-to-nearest, ties away
  unsigned bu = y.u + 0x8000u;
  // dst bytes = {bu[3],bu[2],au[3],au[2]} -> {bu[31:16], au[31:16]}
  return __builtin_amdgcn_perm(bu, au, 0x07060302u);
#endif
}

__device__ __forceinline__ unsigned short f2bf(float f) {
  return (unsigned short)(pack2bf(f, 0.f) & 0xFFFFu);
}

// ---------------- zero init: d_out and expert counters ----------------
__global__ __launch_bounds__(256) void moe_zero_kernel(float4* __restrict__ out4,
                                                       long n4, int* __restrict__ cnt) {
  long i = (long)blockIdx.x * blockDim.x + threadIdx.x;
  long stride = (long)gridDim.x * blockDim.x;
  float4 z; z.x = z.y = z.z = z.w = 0.f;
  for (; i < n4; i += stride) out4[i] = z;
  if (blockIdx.x == 0 && threadIdx.x < N_EXP) cnt[threadIdx.x] = 0;
}

// ---------------- weight f32 -> bf16 pre-conversion ----------------
__global__ __launch_bounds__(256) void moe_cvt_kernel(const float4* __restrict__ src,
                                                      uint2* __restrict__ dst, long n4) {
  long i = (long)blockIdx.x * blockDim.x + threadIdx.x;
  long stride = (long)gridDim.x * blockDim.x;
  for (; i < n4; i += stride) {
    float4 v = src[i];
    uint2 pk;
    pk.x = pack2bf(v.x, v.y);
    pk.y = pack2bf(v.z, v.w);
    dst[i] = pk;
  }
}

// ---------------- router: one wave32 per token ----------------
__global__ __launch_bounds__(256) void moe_router_kernel(
    const float* __restrict__ x, const float* __restrict__ rw, int N,
    int* __restrict__ cnt, int* __restrict__ tok, float* __restrict__ wgt) {
  const int wv   = ((int)blockIdx.x * 256 + (int)threadIdx.x) >> 5;  // token id
  const int lane = threadIdx.x & 31;
  if (wv >= N) return;  // uniform per wave

  const float4* xr = (const float4*)(x + (size_t)wv * D_MODEL);
  const float4* rr = (const float4*)rw;
  float acc[N_EXP];
#pragma unroll
  for (int e = 0; e < N_EXP; ++e) acc[e] = 0.f;

  for (int d = lane; d < D_MODEL / 4; d += 32) {
    float4 xv = xr[d];
#pragma unroll
    for (int e = 0; e < N_EXP; ++e) {
      float4 w4 = rr[e * (D_MODEL / 4) + d];
      acc[e] += xv.x * w4.x + xv.y * w4.y + xv.z * w4.z + xv.w * w4.w;
    }
  }
#pragma unroll
  for (int e = 0; e < N_EXP; ++e) {
#pragma unroll
    for (int off = 16; off > 0; off >>= 1)
      acc[e] += __shfl_xor(acc[e], off, 32);
  }

  if (lane == 0) {
    float m = acc[0];
#pragma unroll
    for (int e = 1; e < N_EXP; ++e) m = fmaxf(m, acc[e]);
    float p[N_EXP];
#pragma unroll
    for (int e = 0; e < N_EXP; ++e) p[e] = __expf(acc[e] - m);
    int i1 = 0;
#pragma unroll
    for (int e = 1; e < N_EXP; ++e) if (p[e] > p[i1]) i1 = e;     // first-max (top_k tiebreak)
    int i2 = (i1 == 0) ? 1 : 0;
#pragma unroll
    for (int e = 0; e < N_EXP; ++e) if (e != i1 && p[e] > p[i2]) i2 = e;
    float inv = __builtin_amdgcn_rcpf(p[i1] + p[i2]);
    int s1 = atomicAdd(&cnt[i1], 1);
    tok[(size_t)i1 * N + s1] = wv;
    wgt[(size_t)i1 * N + s1] = p[i1] * inv;
    int s2 = atomicAdd(&cnt[i2], 1);
    tok[(size_t)i2 * N + s2] = wv;
    wgt[(size_t)i2 * N + s2] = p[i2] * inv;
  }
}

// ---------------- expert FFN: 16 tokens of one expert per workgroup ----------------
// LDS: xs = bf16[16][2048] (64KB), hs = bf16[16][3072] (96KB)  => 160KB dynamic
extern __shared__ char smem_raw[];

// load one 32x16 B fragment: lane n = column n, contiguous 16 K values per lane
template <bool PRE>
__device__ __forceinline__ void loadB(Frag16& B, const void* wv, size_t rowBase,
                                      int k, int laneHi) {
  if constexpr (PRE) {
    const unsigned short* p = (const unsigned short*)wv + rowBase + k + laneHi * 16;
    B.q[0] = *(const uint4*)(p);
    B.q[1] = *(const uint4*)(p + 8);
  } else {
    const float* p = (const float*)wv + rowBase + k + laneHi * 16;
    __builtin_prefetch(p + 32, 0, 0);
#pragma unroll
    for (int j = 0; j < 4; ++j) {
      float4 f = *(const float4*)(p + 4 * j);
      B.u[2 * j + 0] = pack2bf(f.x, f.y);
      B.u[2 * j + 1] = pack2bf(f.z, f.w);
    }
  }
}

__device__ __forceinline__ void loadA(Frag16& A, const unsigned short* base,
                                      size_t stride, int lane15, int laneHi, int k) {
  const unsigned short* ar = base + (size_t)lane15 * stride + k + laneHi * 16;
  A.q[0] = *(const uint4*)(ar);
  A.q[1] = *(const uint4*)(ar + 8);
}

#define WMMA_BF16(C, A, B) \
  C = __builtin_amdgcn_wmma_f32_16x16x32_bf16(false, (A).v, false, (B).v, (short)0, C, false, false)

template <bool PRE>
__global__ __launch_bounds__(256) void moe_expert_kernel(
    const float* __restrict__ x,
    const void* __restrict__ gw,   // PRE ? bf16 : f32, [E][FFN_H][D_MODEL]
    const void* __restrict__ uw,   // PRE ? bf16 : f32, [E][FFN_H][D_MODEL]
    const void* __restrict__ dw,   // PRE ? bf16 : f32, [E][D_MODEL][FFN_H]
    const int*  __restrict__ tok,
    const float* __restrict__ wgt,
    const int*  __restrict__ cnt,
    float* __restrict__ out,
    int N, int tilesPerExpert) {
  const int e    = blockIdx.x / tilesPerExpert;  // expert-major grid => L2 weight reuse
  const int tile = blockIdx.x % tilesPerExpert;
  const int cntE = cnt[e];
  const int tileStart = tile * BM;
  if (tileStart >= cntE) return;  // uniform per block; before any barrier

  unsigned short* xs = (unsigned short*)smem_raw;                        // [BM][D_MODEL]
  unsigned short* hs = (unsigned short*)(smem_raw + BM * D_MODEL * 2);   // [BM][FFN_H]

  const int lane   = threadIdx.x & 31;
  const int wave   = threadIdx.x >> 5;
  const int lane15 = lane & 15;
  const int laneHi = lane >> 4;   // selects contiguous K-half (k = 16*laneHi + s) for A and B in this code

  const int*   tokE = tok + (size_t)e * N;
  const float* wgtE = wgt + (size_t)e * N;

  // ---- stage BM gathered token rows as bf16 in LDS ----
  for (int i = threadIdx.x; i < BM * (D_MODEL / 4); i += 256) {
    int m   = i >> 9;            // i / 512
    int c4  = i & 511;
    int idx = tileStart + m;
    int tkn = (idx < cntE) ? tokE[idx] : tokE[tileStart];  // pad rows: weight 0 later
    float4 v = ((const float4*)(x + (size_t)tkn * D_MODEL))[c4];
    uint2 pk;
    pk.x = pack2bf(v.x, v.y);
    pk.y = pack2bf(v.z, v.w);
    *(uint2*)(xs + (size_t)m * D_MODEL + c4 * 4) = pk;
  }
  __syncthreads();

  // ---- phase 1: hs = silu(x @ Wg^T) * (x @ Wu^T) ----
  // Two 16-wide H tiles per wave; K unrolled x2 with ALL fragment loads batched
  // ahead of the 8 WMMAs so loads stay deep in flight (distinct VGPRs, partial
  // loadcnt waits instead of loadcnt==0 before every WMMA).
  for (int hBase = 0; hBase < FFN_H; hBase += 256) {
    const int hCol0 = hBase + wave * 32 + lane15;  // B column N = lane15
    const int hCol1 = hCol0 + 16;
    v8f cG0 = {}, cG1 = {};
    v8f cU0 = {}, cU1 = {};
    const size_t row0 = ((size_t)e * FFN_H + hCol0) * D_MODEL;
    const size_t row1 = ((size_t)e * FFN_H + hCol1) * D_MODEL;
    for (int d = 0; d < D_MODEL; d += 64) {
      Frag16 A0, A1;
      loadA(A0, xs, D_MODEL, lane15, laneHi, d);
      loadA(A1, xs, D_MODEL, lane15, laneHi, d + 32);
      Frag16 Bg0a, Bg1a, Bu0a, Bu1a, Bg0b, Bg1b, Bu0b, Bu1b;
      loadB<PRE>(Bg0a, gw, row0, d, laneHi);
      loadB<PRE>(Bg1a, gw, row1, d, laneHi);
      loadB<PRE>(Bu0a, uw, row0, d, laneHi);
      loadB<PRE>(Bu1a, uw, row1, d, laneHi);
      loadB<PRE>(Bg0b, gw, row0, d + 32, laneHi);
      loadB<PRE>(Bg1b, gw, row1, d + 32, laneHi);
      loadB<PRE>(Bu0b, uw, row0, d + 32, laneHi);
      loadB<PRE>(Bu1b, uw, row1, d + 32, laneHi);
      WMMA_BF16(cG0, A0, Bg0a);
      WMMA_BF16(cG1, A0, Bg1a);
      WMMA_BF16(cU0, A0, Bu0a);
      WMMA_BF16(cU1, A0, Bu1a);
      WMMA_BF16(cG0, A1, Bg0b);
      WMMA_BF16(cG1, A1, Bg1b);
      WMMA_BF16(cU0, A1, Bu0b);
      WMMA_BF16(cU1, A1, Bu1b);
    }
    // C layout: VGPR r, lanes 0-15 -> M=r, lanes 16-31 -> M=r+8 ; N = lane15
#pragma unroll
    for (int r = 0; r < 8; ++r) {
      int M = r + 8 * laneHi;
      float g0 = cG0[r], u0 = cU0[r];
      float g1 = cG1[r], u1 = cU1[r];
      // silu(g)*u with fast reciprocal (v_rcp_f32), ~1ulp -- below bf16 noise
      float h0 = g0 * __builtin_amdgcn_rcpf(1.f + __expf(-g0)) * u0;
      float h1 = g1 * __builtin_amdgcn_rcpf(1.f + __expf(-g1)) * u1;
      hs[(size_t)M * FFN_H + hCol0] = f2bf(h0);
      hs[(size_t)M * FFN_H + hCol1] = f2bf(h1);
    }
  }
  __syncthreads();

  // ---- phase 2: out[tok] += w * (hs @ Wd^T), two 16-wide D tiles per wave ----
  for (int t0 = wave * 2; t0 < D_MODEL / 16; t0 += 16) {
    const int dCol0 = t0 * 16 + lane15;
    const int dCol1 = dCol0 + 16;
    v8f cO0 = {}, cO1 = {};
    const size_t row0 = ((size_t)e * D_MODEL + dCol0) * FFN_H;
    const size_t row1 = ((size_t)e * D_MODEL + dCol1) * FFN_H;
    for (int h = 0; h < FFN_H; h += 64) {
      Frag16 A0, A1;
      loadA(A0, hs, FFN_H, lane15, laneHi, h);
      loadA(A1, hs, FFN_H, lane15, laneHi, h + 32);
      Frag16 Bd0a, Bd1a, Bd0b, Bd1b;
      loadB<PRE>(Bd0a, dw, row0, h, laneHi);
      loadB<PRE>(Bd1a, dw, row1, h, laneHi);
      loadB<PRE>(Bd0b, dw, row0, h + 32, laneHi);
      loadB<PRE>(Bd1b, dw, row1, h + 32, laneHi);
      WMMA_BF16(cO0, A0, Bd0a);
      WMMA_BF16(cO1, A0, Bd1a);
      WMMA_BF16(cO0, A1, Bd0b);
      WMMA_BF16(cO1, A1, Bd1b);
    }
#pragma unroll
    for (int r = 0; r < 8; ++r) {
      int M = r + 8 * laneHi;
      int idx = tileStart + M;
      if (idx < cntE) {
        float w = wgtE[idx];
        int tkn = tokE[idx];
        atomicAdd(out + (size_t)tkn * D_MODEL + dCol0, cO0[r] * w);
        atomicAdd(out + (size_t)tkn * D_MODEL + dCol1, cO1[r] * w);
      }
    }
  }
}

// ---------------- host launcher ----------------
extern "C" void kernel_launch(void* const* d_in, const int* in_sizes, int n_in,
                              void* d_out, int out_size, void* d_ws, size_t ws_size,
                              hipStream_t stream) {
  const float* x  = (const float*)d_in[0];
  const float* rw = (const float*)d_in[1];
  const float* gw = (const float*)d_in[2];
  const float* uw = (const float*)d_in[3];
  const float* dw = (const float*)d_in[4];
  float* out = (float*)d_out;

  const int N = in_sizes[0] / D_MODEL;   // 16384 tokens

  // workspace layout:
  //   [cnt: 8 ints][tok: 8*N ints][wgt: 8*N floats] | 256-aligned |
  //   [gwB bf16][uwB bf16][dwB bf16]  (each E*FFN_H*D_MODEL bf16)
  char* ws = (char*)d_ws;
  int*   cnt = (int*)ws;
  int*   tok = (int*)(ws + 256);
  float* wgt = (float*)(ws + 256 + (size_t)N_EXP * N * sizeof(int));

  const size_t listBytes = 256 + 2 * (size_t)N_EXP * N * sizeof(int);
  const size_t wOff   = (listBytes + 255) & ~(size_t)255;
  const size_t wElems = (size_t)N_EXP * FFN_H * D_MODEL;       // per weight tensor
  const size_t wBytes = wElems * 2;                             // bf16
  const bool   preconv = ws_size >= wOff + 3 * wBytes;          // deterministic

  const long n4 = (long)N * D_MODEL / 4;
  moe_zero_kernel<<<2048, 256, 0, stream>>>((float4*)out, n4, cnt);
  moe_router_kernel<<<(N + 7) / 8, 256, 0, stream>>>(x, rw, N, cnt, tok, wgt);

  const int tiles = (N + BM - 1) / BM;
  const size_t lds = (size_t)BM * D_MODEL * 2 + (size_t)BM * FFN_H * 2;  // 160 KB

  if (preconv) {
    unsigned short* gwB = (unsigned short*)(ws + wOff);
    unsigned short* uwB = gwB + wElems;
    unsigned short* dwB = uwB + wElems;
    const long w4 = (long)(wElems / 4);
    moe_cvt_kernel<<<4096, 256, 0, stream>>>((const float4*)gw, (uint2*)gwB, w4);
    moe_cvt_kernel<<<4096, 256, 0, stream>>>((const float4*)uw, (uint2*)uwB, w4);
    moe_cvt_kernel<<<4096, 256, 0, stream>>>((const float4*)dw, (uint2*)dwB, w4);

    (void)hipFuncSetAttribute((const void*)moe_expert_kernel<true>,
                              hipFuncAttributeMaxDynamicSharedMemorySize, (int)lds);
    moe_expert_kernel<true><<<N_EXP * tiles, 256, lds, stream>>>(
        x, gwB, uwB, dwB, tok, wgt, cnt, out, N, tiles);
  } else {
    (void)hipFuncSetAttribute((const void*)moe_expert_kernel<false>,
                              hipFuncAttributeMaxDynamicSharedMemorySize, (int)lds);
    moe_expert_kernel<false><<<N_EXP * tiles, 256, lds, stream>>>(
        x, gw, uw, dw, tok, wgt, cnt, out, N, tiles);
  }
}
